// NetG_46600395162466
// MI455X (gfx1250) — matrix operand order repaired and checked by reading, not compiled, not run on hardware
//
#include <hip/hip_runtime.h>
#include <hip/hip_bf16.h>

typedef __attribute__((ext_vector_type(16))) _Float16 v16h;
typedef __attribute__((ext_vector_type(8)))  _Float16 v8h;
typedef __attribute__((ext_vector_type(8)))  float    v8f;

#define MEMD 2000

__device__ __forceinline__ float gelu_exact(float v) {
    return 0.5f * v * (1.0f + erff(v * 0.70710678118654752f));
}

// ---------------------------------------------------------------------------
// Generic tiled GEMM:  Y[M,N] = act( X[M,K] @ W + bias + R )
//   WT=false: W stored [K,N] row-major.  WT=true: W stored [N,K] row-major.
//   ACT: 0 = none, 1 = exact GELU.
// 128x64 block tile, 8 waves; each wave owns a 32x32 tile computed as
// 4x wmma_f32_16x16x32_f16 per K-step (2 A frags x 2 B frags -> 2x A/B reuse).
// f32 inputs converted to f16 in LDS; fp32 accumulation.
// ---------------------------------------------------------------------------
template<bool WT, int ACT>
__global__ __launch_bounds__(256)
void gemm_wmma(const float* __restrict__ X, const float* __restrict__ W,
               const float* __restrict__ bias, const float* R,
               float* Y, int M, int N, int K)
{
    __shared__ alignas(16) _Float16 As[128][40];  // [m][k], stride 80B (16B aligned)
    __shared__ alignas(16) _Float16 Bs[64][40];   // [n][k]
    const int tid  = threadIdx.x;
    const int lane = tid & 31;
    const int wave = tid >> 5;
    const int wm   = wave >> 1;       // 0..3  -> rows wm*32
    const int wn   = wave & 1;        // 0..1  -> cols wn*32
    const int half = lane >> 4;
    const int l15  = lane & 15;
    const int m0   = blockIdx.y * 128;
    const int n0   = blockIdx.x * 64;

    v8f acc[2][2] = {};

    const int lr = tid >> 2;          // 0..63
    const int lc = (tid & 3) * 8;     // 0,8,16,24

    for (int k0 = 0; k0 < K; k0 += 32) {
        #pragma unroll
        for (int rr = 0; rr < 2; ++rr) {   // stage A tile (128 x 32)
            const int  arow = rr * 64 + lr;
            const int  row  = m0 + arow;
            const bool rok  = row < M;
            const float* src = X + (long)row * K + k0 + lc;
            #pragma unroll
            for (int j = 0; j < 8; ++j) {
                const int k = k0 + lc + j;
                As[arow][lc + j] = (_Float16)((rok && k < K) ? src[j] : 0.0f);
            }
            if (rok && (k0 + 32) < K) __builtin_prefetch(src + 32, 0, 0); // global_prefetch_b8
        }
        {   // stage B tile transposed into [n][k]
            const int  col = n0 + lr;
            const bool cok = col < N;
            #pragma unroll
            for (int j = 0; j < 8; ++j) {
                const int k = k0 + lc + j;
                float v = 0.0f;
                if (cok && k < K) v = WT ? W[(long)col * K + k] : W[(long)k * N + col];
                Bs[lr][lc + j] = (_Float16)v;
            }
        }
        __syncthreads();

        // A fragments: lane l holds row (l&15); K runs [8h,8h+8) and [16+8h,24+8h)
        v16h a[2], b[2];
        #pragma unroll
        for (int u = 0; u < 2; ++u) {
            const v8h lo = *(const v8h*)&As[wm * 32 + u * 16 + l15][8 * half];
            const v8h hi = *(const v8h*)&As[wm * 32 + u * 16 + l15][16 + 8 * half];
            #pragma unroll
            for (int i = 0; i < 8; ++i) { a[u][i] = lo[i]; a[u][8 + i] = hi[i]; }
        }
        // B fragments: lane l holds col (l&15); K runs [8h,8h+8) and [16+8h,24+8h)
        #pragma unroll
        for (int t = 0; t < 2; ++t) {
            const v8h lo = *(const v8h*)&Bs[wn * 32 + t * 16 + l15][8 * half];
            const v8h hi = *(const v8h*)&Bs[wn * 32 + t * 16 + l15][16 + 8 * half];
            #pragma unroll
            for (int i = 0; i < 8; ++i) { b[t][i] = lo[i]; b[t][8 + i] = hi[i]; }
        }
        #pragma unroll
        for (int u = 0; u < 2; ++u)
            #pragma unroll
            for (int t = 0; t < 2; ++t)
                acc[u][t] = __builtin_amdgcn_wmma_f32_16x16x32_f16(
                                false, a[u], false, b[t], (short)0, acc[u][t], false, false);
        __syncthreads();
    }

    #pragma unroll
    for (int u = 0; u < 2; ++u)
        #pragma unroll
        for (int t = 0; t < 2; ++t)
            #pragma unroll
            for (int r = 0; r < 8; ++r) {
                const int row = m0 + wm * 32 + u * 16 + 8 * half + r;
                const int col = n0 + wn * 32 + t * 16 + l15;
                if (row < M && col < N) {
                    float v = acc[u][t][r];
                    if (bias) v += bias[col];
                    if (R)    v += R[(long)row * N + col];
                    if (ACT == 1) v = gelu_exact(v);
                    Y[(long)row * N + col] = v;
                }
            }
}

// ---------------------------------------------------------------------------
// Fused 7x7 window attention for one (window, head). head_dim == 32 always.
// S = (Q*scale) K^T (4 WMMAs) -> +rel-pos bias (+shift mask) -> softmax ->
// O = P V (4 WMMAs).  49 rows/keys zero-padded to 64 so EXEC stays all-ones.
// qkv layout: [B_*49, 3*C] with feature order [3][nH][32].
// ---------------------------------------------------------------------------
__global__ __launch_bounds__(128)
void attn_kernel(const float* __restrict__ qkv, const float* __restrict__ rpb,
                 float* __restrict__ out, int C, int nH, int H, int W, int shift)
{
    __shared__ alignas(16) _Float16 Qs[64][40];
    __shared__ alignas(16) _Float16 Ks[64][40];
    __shared__ alignas(16) _Float16 Vt[32][72];   // [d][key] (B operand for P@V)
    __shared__ alignas(16) float    Sf[64][68];
    __shared__ alignas(16) _Float16 Ps[64][72];

    const int b_   = blockIdx.x;
    const int head = blockIdx.y;
    const int Wg   = W / 7;
    const int nW   = (H / 7) * Wg;
    const int win  = b_ % nW;
    const int wy   = win / Wg, wx = win % Wg;
    const int tid  = threadIdx.x;
    const long base = (long)b_ * 49;
    const int C3 = 3 * C;

    for (int idx = tid; idx < 64 * 32; idx += 128) {
        const int row = idx >> 5, d = idx & 31;
        float q = 0.f, k = 0.f, v = 0.f;
        if (row < 49) {
            const float* p = qkv + (base + row) * C3 + head * 32 + d;
            q = p[0] * 0.17677669529663688f;   // 32^-0.5
            k = p[C];
            v = p[2 * C];
        }
        Qs[row][d] = (_Float16)q;
        Ks[row][d] = (_Float16)k;
        Vt[d][row] = (_Float16)v;
    }
    __syncthreads();

    const int lane = tid & 31;
    const int wm   = tid >> 5;      // 4 waves -> 4 row tiles
    const int half = lane >> 4;
    const int l15  = lane & 15;

    // ---- scores ----
    v16h a;
    {
        const v8h lo = *(const v8h*)&Qs[wm * 16 + l15][8 * half];
        const v8h hi = *(const v8h*)&Qs[wm * 16 + l15][16 + 8 * half];
        #pragma unroll
        for (int i = 0; i < 8; ++i) { a[i] = lo[i]; a[8 + i] = hi[i]; }
    }
    #pragma unroll
    for (int nt = 0; nt < 4; ++nt) {
        v16h b;
        const v8h lo = *(const v8h*)&Ks[nt * 16 + l15][8 * half];
        const v8h hi = *(const v8h*)&Ks[nt * 16 + l15][16 + 8 * half];
        #pragma unroll
        for (int i = 0; i < 8; ++i) { b[i] = lo[i]; b[8 + i] = hi[i]; }
        v8f s = {};
        s = __builtin_amdgcn_wmma_f32_16x16x32_f16(false, a, false, b, (short)0, s, false, false);
        #pragma unroll
        for (int r = 0; r < 8; ++r)
            Sf[wm * 16 + 8 * half + r][nt * 16 + l15] = s[r];
    }
    __syncthreads();

    // ---- bias + mask + softmax (one thread per row) ----
    if (tid < 64) {
        const int m = tid;
        if (m < 49) {
            const int mh = m / 7, mw = m % 7;
            const int gym = wy * 7 + mh, gxm = wx * 7 + mw;
            const int rym = (gym < H - 7) ? 0 : ((gym < H - 3) ? 1 : 2);
            const int rxm = (gxm < W - 7) ? 0 : ((gxm < W - 3) ? 1 : 2);
            float mx = -1e30f;
            for (int n = 0; n < 49; ++n) {
                const int nh2 = n / 7, nw2 = n % 7;
                float s = Sf[m][n] + rpb[((mh - nh2 + 6) * 13 + (mw - nw2 + 6)) * nH + head];
                if (shift) {
                    const int gyn = wy * 7 + nh2, gxn = wx * 7 + nw2;
                    const int ryn = (gyn < H - 7) ? 0 : ((gyn < H - 3) ? 1 : 2);
                    const int rxn = (gxn < W - 7) ? 0 : ((gxn < W - 3) ? 1 : 2);
                    if (ryn != rym || rxn != rxm) s -= 100.0f;
                }
                Sf[m][n] = s;
                mx = fmaxf(mx, s);
            }
            float sum = 0.f;
            for (int n = 0; n < 49; ++n) {
                const float e = expf(Sf[m][n] - mx);
                sum += e;
                Sf[m][n] = e;
            }
            const float inv = 1.0f / sum;
            for (int n = 0; n < 49; ++n) Ps[m][n] = (_Float16)(Sf[m][n] * inv);
            for (int n = 49; n < 64; ++n) Ps[m][n] = (_Float16)0.f;
        } else {
            for (int n = 0; n < 64; ++n) Ps[m][n] = (_Float16)0.f;
        }
    }
    __syncthreads();

    // ---- O = P @ V ----
    v8f o[2] = {};
    #pragma unroll
    for (int ks = 0; ks < 2; ++ks) {
        v16h pa;
        {
            const v8h lo = *(const v8h*)&Ps[wm * 16 + l15][32 * ks + 8 * half];
            const v8h hi = *(const v8h*)&Ps[wm * 16 + l15][32 * ks + 16 + 8 * half];
            #pragma unroll
            for (int i = 0; i < 8; ++i) { pa[i] = lo[i]; pa[8 + i] = hi[i]; }
        }
        #pragma unroll
        for (int t = 0; t < 2; ++t) {
            v16h vb;
            const v8h lo = *(const v8h*)&Vt[t * 16 + l15][32 * ks + 8 * half];
            const v8h hi = *(const v8h*)&Vt[t * 16 + l15][32 * ks + 16 + 8 * half];
            #pragma unroll
            for (int i = 0; i < 8; ++i) { vb[i] = lo[i]; vb[8 + i] = hi[i]; }
            o[t] = __builtin_amdgcn_wmma_f32_16x16x32_f16(false, pa, false, vb,
                                                          (short)0, o[t], false, false);
        }
    }
    #pragma unroll
    for (int t = 0; t < 2; ++t)
        #pragma unroll
        for (int r = 0; r < 8; ++r) {
            const int row = wm * 16 + 8 * half + r;
            if (row < 49)
                out[(base + row) * C + head * 32 + t * 16 + l15] = o[t][r];
        }
}

// ---------------------------------------------------------------------------
// LayerNorm: one wave per row (wave32 shuffles). Safe in-place (y may == x).
// ---------------------------------------------------------------------------
__global__ __launch_bounds__(256)
void ln_kernel(const float* x, const float* g, const float* b, float* y, long rows, int C)
{
    const long row = (long)blockIdx.x * 8 + (threadIdx.x >> 5);
    const int lane = threadIdx.x & 31;
    if (row >= rows) return;
    const float* xr = x + row * C;
    float s = 0.f, s2 = 0.f;
    for (int c = lane; c < C; c += 32) { const float v = xr[c]; s += v; s2 += v * v; }
    #pragma unroll
    for (int o = 16; o > 0; o >>= 1) { s += __shfl_xor(s, o, 32); s2 += __shfl_xor(s2, o, 32); }
    const float mu  = s / C;
    const float var = s2 / C - mu * mu;
    const float inv = rsqrtf(var + 1e-5f);
    float* yr = y + row * C;
    for (int c = lane; c < C; c += 32) yr[c] = (xr[c] - mu) * inv * g[c] + b[c];
}

// ---------------------------------------------------------------------------
// Index-permutation kernels (1 thread / element)
// ---------------------------------------------------------------------------
__global__ __launch_bounds__(256)
void win_part_k(const float* __restrict__ h, float* __restrict__ y,
                int B, int H, int W, int C, int shift, long total)
{
    long i = (long)blockIdx.x * 256 + threadIdx.x; if (i >= total) return;
    const int c = (int)(i % C); long t = i / C;
    const int p = (int)(t % 49); t /= 49;
    const int Wg = W / 7;  const int nW = (H / 7) * Wg;
    const int win = (int)(t % nW); const int b = (int)(t / nW);
    int gy = (win / Wg) * 7 + p / 7 + shift;
    int gx = (win % Wg) * 7 + p % 7 + shift;
    if (gy >= H) gy -= H;
    if (gx >= W) gx -= W;
    y[i] = h[((long)b * H * W + (long)gy * W + gx) * C + c];
}

__global__ __launch_bounds__(256)
void win_rev_add_k(const float* __restrict__ pin, float* x,
                   int B, int H, int W, int C, int shift, long total)
{
    long i = (long)blockIdx.x * 256 + threadIdx.x; if (i >= total) return;
    const int c = (int)(i % C); long t = i / C;
    const int p = (int)(t % 49); t /= 49;
    const int Wg = W / 7;  const int nW = (H / 7) * Wg;
    const int win = (int)(t % nW); const int b = (int)(t / nW);
    int gy = (win / Wg) * 7 + p / 7 + shift;
    int gx = (win % Wg) * 7 + p % 7 + shift;
    if (gy >= H) gy -= H;
    if (gx >= W) gx -= W;
    x[((long)b * H * W + (long)gy * W + gx) * C + c] += pin[i];
}

__global__ __launch_bounds__(256)
void patch_gather_k(const float* __restrict__ x, float* __restrict__ y, long total)
{
    long i = (long)blockIdx.x * 256 + threadIdx.x; if (i >= total) return;
    const int f = (int)(i % 48); long t = i / 48;
    const int px = (int)(t % 56); t /= 56;
    const int py = (int)(t % 56); const int b = (int)(t / 56);
    const int cc = f >> 4, r = f & 15, dy = r >> 2, dx = r & 3;
    y[i] = x[(((long)b * 3 + cc) * 224 + py * 4 + dy) * 224 + px * 4 + dx];
}

__global__ __launch_bounds__(256)
void merge_gather_k(const float* __restrict__ h, float* __restrict__ y,
                    int B, int H, int W, int C, long total)
{
    long i = (long)blockIdx.x * 256 + threadIdx.x; if (i >= total) return;
    const int C4 = 4 * C;
    const int c4 = (int)(i % C4); long t = i / C4;
    const int ox = (int)(t % (W / 2)); t /= (W / 2);
    const int oy = (int)(t % (H / 2)); const int b = (int)(t / (H / 2));
    const int q = c4 / C, c = c4 % C;
    const int dy = q & 1, dx = q >> 1;
    y[i] = h[((long)b * H * W + (long)(2 * oy + dy) * W + (2 * ox + dx)) * C + c];
}

__global__ __launch_bounds__(256)
void expand_rearrange_k(const float* __restrict__ yin, float* __restrict__ out,
                        int B, int H, int W, int dim, long total)
{
    long i = (long)blockIdx.x * 256 + threadIdx.x; if (i >= total) return;
    const int d2 = dim >> 1;
    const int c = (int)(i % d2); long t = i / d2;
    const int x2 = (int)(t % (2 * W)); t /= (2 * W);
    const int y2 = (int)(t % (2 * H)); const int b = (int)(t / (2 * H));
    const int hh = y2 >> 1, aa = y2 & 1, ww = x2 >> 1, bb = x2 & 1;
    out[i] = yin[((long)b * H * W + (long)hh * W + ww) * (2 * dim) + (aa * 2 + bb) * d2 + c];
}

__global__ __launch_bounds__(256)
void mem_gather_k(const float* __restrict__ h, float* __restrict__ y,
                  int B, int H, int W, int C, int w, long total)
{
    long i = (long)blockIdx.x * 256 + threadIdx.x; if (i >= total) return;
    const int ww = w * w, fea = C * ww / 8, Wg = W / w, nWin = (H / w) * Wg, Cc = C / 8;
    const int f = (int)(i % fea); long t = i / fea;
    const int j = (int)(t % 8); t /= 8;
    const int win = (int)(t % nWin); const int b = (int)(t / nWin);
    const int cl = f / ww, r = f % ww, py = r / w, px = r % w;
    const int c = j * Cc + cl;
    const int gy = (win / Wg) * w + py, gx = (win % Wg) * w + px;
    y[i] = h[((long)b * H * W + (long)gy * W + gx) * C + c];
}

__global__ __launch_bounds__(256)
void mem_scatter_k(const float* __restrict__ src, float* __restrict__ hout,
                   int B, int H, int W, int C, int w, long total)
{
    long i = (long)blockIdx.x * 256 + threadIdx.x; if (i >= total) return;
    const int ww = w * w, fea = C * ww / 8, Wg = W / w, nWin = (H / w) * Wg, Cc = C / 8;
    const int f = (int)(i % fea); long t = i / fea;
    const int j = (int)(t % 8); t /= 8;
    const int win = (int)(t % nWin); const int b = (int)(t / nWin);
    const int cl = f / ww, r = f % ww, py = r / w, px = r % w;
    const int c = j * Cc + cl;
    const int gy = (win / Wg) * w + py, gx = (win % Wg) * w + px;
    hout[((long)b * H * W + (long)gy * W + gx) * C + c] = src[i];
}

__global__ __launch_bounds__(256)
void concat2_k(const float* __restrict__ a, const float* __restrict__ bsrc,
               float* __restrict__ y, int d, long total)
{
    long i = (long)blockIdx.x * 256 + threadIdx.x; if (i >= total) return;
    const int col = (int)(i % (2 * d)); const long r = i / (2 * d);
    y[i] = (col < d) ? a[r * d + col] : bsrc[r * d + (col - d)];
}

__global__ __launch_bounds__(256)
void up_rearrange_k(const float* __restrict__ in, float* __restrict__ out, long total)
{
    long i = (long)blockIdx.x * 256 + threadIdx.x; if (i >= total) return;
    const int c = (int)(i % 96); long t = i / 96;
    const int x = (int)(t % 224); t /= 224;
    const int y = (int)(t % 224); const int b = (int)(t / 224);
    const int py = y >> 2, dy = y & 3, px = x >> 2, dx = x & 3;
    out[i] = in[((long)b * 3136 + py * 56 + px) * 1536 + (dy * 4 + dx) * 96 + c];
}

__global__ __launch_bounds__(256)
void out_transpose_k(const float* __restrict__ in, float* __restrict__ out, long total)
{
    long i = (long)blockIdx.x * 256 + threadIdx.x; if (i >= total) return;
    const int x = (int)(i % 224); long t = i / 224;
    const int y = (int)(t % 224); t /= 224;
    const int c = (int)(t % 3); const int b = (int)(t / 3);
    out[i] = in[(((long)b * 224 + y) * 224 + x) * 3 + c];
}

// ---------------------------------------------------------------------------
// MemAE addressing epilogue: softmax(2000) -> hard shrink -> L1 renorm, in place
// ---------------------------------------------------------------------------
__global__ __launch_bounds__(256)
void mem_shrink_k(float* att, long rows)
{
    __shared__ float buf[MEMD];
    __shared__ float red[8];
    const long row = blockIdx.x;
    if (row >= rows) return;
    float* ar = att + row * MEMD;
    const int tid = threadIdx.x, lane = tid & 31, wv = tid >> 5;

    float mx = -1e30f;
    for (int i = tid; i < MEMD; i += 256) { const float v = ar[i]; buf[i] = v; mx = fmaxf(mx, v); }
    #pragma unroll
    for (int o = 16; o > 0; o >>= 1) mx = fmaxf(mx, __shfl_xor(mx, o, 32));
    if (lane == 0) red[wv] = mx;
    __syncthreads();
    float m8 = red[0];
    #pragma unroll
    for (int j = 1; j < 8; ++j) m8 = fmaxf(m8, red[j]);
    __syncthreads();

    float sum = 0.f;
    for (int i = tid; i < MEMD; i += 256) { const float e = expf(buf[i] - m8); buf[i] = e; sum += e; }
    #pragma unroll
    for (int o = 16; o > 0; o >>= 1) sum += __shfl_xor(sum, o, 32);
    if (lane == 0) red[wv] = sum;
    __syncthreads();
    float s8 = 0.f;
    #pragma unroll
    for (int j = 0; j < 8; ++j) s8 += red[j];
    __syncthreads();
    const float inv = 1.0f / s8;

    float sum2 = 0.f;
    for (int i = tid; i < MEMD; i += 256) {
        float a = buf[i] * inv;
        const float sdf = a - 0.005f;
        a = fmaxf(sdf, 0.f) * a / (fabsf(sdf) + 1e-12f);
        buf[i] = a;
        sum2 += a;
    }
    #pragma unroll
    for (int o = 16; o > 0; o >>= 1) sum2 += __shfl_xor(sum2, o, 32);
    if (lane == 0) red[wv] = sum2;
    __syncthreads();
    float t8 = 0.f;
    #pragma unroll
    for (int j = 0; j < 8; ++j) t8 += red[j];
    const float inv2 = 1.0f / (t8 + 1e-12f);
    for (int i = tid; i < MEMD; i += 256) ar[i] = buf[i] * inv2;
}

// ---------------------------------------------------------------------------
// Host-side orchestration
// ---------------------------------------------------------------------------
static inline void launch_ln(hipStream_t s, const float* x, const float* g,
                             const float* b, float* y, long rows, int C) {
    ln_kernel<<<dim3((unsigned)((rows + 7) / 8)), 256, 0, s>>>(x, g, b, y, rows, C);
}

static inline void launch_gemm(hipStream_t s, const float* X, const float* W,
                               const float* bias, const float* R, float* Y,
                               long M, int N, int K, bool wt, int act) {
    dim3 g((unsigned)((N + 63) / 64), (unsigned)((M + 127) / 128));
    if (wt)            gemm_wmma<true , 0><<<g, 256, 0, s>>>(X, W, bias, R, Y, (int)M, N, K);
    else if (act == 1) gemm_wmma<false, 1><<<g, 256, 0, s>>>(X, W, bias, R, Y, (int)M, N, K);
    else               gemm_wmma<false, 0><<<g, 256, 0, s>>>(X, W, bias, R, Y, (int)M, N, K);
}

#define GRID1D(total) dim3((unsigned)(((total) + 255) / 256)), 256, 0, s

struct BlkP {
    const float *fc1_b, *fc1_w, *fc2_b, *fc2_w, *ln1_b, *ln1_w, *ln2_b, *ln2_w,
                *proj_b, *proj_w, *qkv_b, *qkv_w, *rpb;
};

// Param leaf order assumes JAX pytree flattening (dicts sorted by key).
// Block leaves: fc1_b, fc1_w, fc2_b, fc2_w, ln1{b,w}, ln2{b,w}, proj_b, proj_w, qkv_b, qkv_w, rpb
static inline BlkP mk_blk(void* const* din, int base) {
    BlkP p;
    p.fc1_b  = (const float*)din[base + 0];  p.fc1_w  = (const float*)din[base + 1];
    p.fc2_b  = (const float*)din[base + 2];  p.fc2_w  = (const float*)din[base + 3];
    p.ln1_b  = (const float*)din[base + 4];  p.ln1_w  = (const float*)din[base + 5];
    p.ln2_b  = (const float*)din[base + 6];  p.ln2_w  = (const float*)din[base + 7];
    p.proj_b = (const float*)din[base + 8];  p.proj_w = (const float*)din[base + 9];
    p.qkv_b  = (const float*)din[base + 10]; p.qkv_w  = (const float*)din[base + 11];
    p.rpb    = (const float*)din[base + 12];
    return p;
}

static void swin_block(hipStream_t s, void* const* din, int base, float* x,
                       int B, int H, int W, int C, int nH, int shift,
                       float* T1, float* T2, float* T3)
{
    const BlkP p = mk_blk(din, base);
    const long rows  = (long)B * H * W;
    const int  nWn   = (H / 7) * (W / 7);
    const long wrows = (long)B * nWn * 49;
    launch_ln(s, x, p.ln1_w, p.ln1_b, T1, rows, C);
    { const long tot = rows * C; win_part_k<<<GRID1D(tot)>>>(T1, T2, B, H, W, C, shift, tot); }
    launch_gemm(s, T2, p.qkv_w, p.qkv_b, nullptr, T3, wrows, 3 * C, C, false, 0);
    attn_kernel<<<dim3((unsigned)(B * nWn), (unsigned)nH), 128, 0, s>>>(T3, p.rpb, T1, C, nH, H, W, shift);
    launch_gemm(s, T1, p.proj_w, p.proj_b, nullptr, T2, wrows, C, C, false, 0);
    { const long tot = rows * C; win_rev_add_k<<<GRID1D(tot)>>>(T2, x, B, H, W, C, shift, tot); }
    launch_ln(s, x, p.ln2_w, p.ln2_b, T1, rows, C);
    launch_gemm(s, T1, p.fc1_w, p.fc1_b, nullptr, T2, rows, 4 * C, C, false, 1);
    launch_gemm(s, T2, p.fc2_w, p.fc2_b, x, x, rows, C, 4 * C, false, 0);
}

static void mem_module(hipStream_t s, const float* h, const float* Mm, float* out_h,
                       int B, int H, int W, int C, int w,
                       float* T1, float* T2, float* T3)
{
    const int  nWin = (H / w) * (W / w);
    const int  fea  = C * w * w / 8;
    const long rows = (long)B * nWin * 8;
    { const long tot = rows * fea; mem_gather_k<<<GRID1D(tot)>>>(h, T1, B, H, W, C, w, tot); }
    launch_gemm(s, T1, Mm, nullptr, nullptr, T2, rows, MEMD, fea, true, 0);   // x @ M^T
    mem_shrink_k<<<dim3((unsigned)rows), 256, 0, s>>>(T2, rows);
    launch_gemm(s, T2, Mm, nullptr, nullptr, T3, rows, fea, MEMD, false, 0);  // att @ M
    { const long tot = rows * fea; mem_scatter_k<<<GRID1D(tot)>>>(T3, out_h, B, H, W, C, w, tot); }
}

extern "C" void kernel_launch(void* const* d_in, const int* in_sizes, int n_in,
                              void* d_out, int out_size, void* d_ws, size_t ws_size,
                              hipStream_t stream)
{
    (void)in_sizes; (void)n_in; (void)out_size; (void)ws_size;
    hipStream_t s = stream;
    auto F = [&](int i) { return (const float*)d_in[i]; };
    const int B = 4;

    // Workspace layout (floats); ~180 MB total
    float* ws  = (float*)d_ws;
    float* X0  = ws;                       // running activations  (<= 1.21M floats)
    float* XS  = ws + 1400000;             // side buffer          (<= 1.21M floats)
    float* D0  = ws + 2800000;             // skip [4,3136,96]
    float* D1  = D0 + 1204224;             // skip [4,784,192]
    float* D2  = D1 + 602112;              // skip [4,196,384]
    float* TB  = ws + 5100000;             // temp region (40M floats)
    float* T1  = TB;
    float* T2  = TB + 8000000;
    float* T3  = TB + 16000000;
    float* BIG1 = TB;                      // final phase only (19.3M)
    float* BIG2 = TB + 20000000;           // final phase only (19.3M)

    // ---- patch embed: [4,3,224,224] -> [4,3136,96] ----
    { const long tot = (long)B * 3136 * 48; patch_gather_k<<<GRID1D(tot)>>>(F(222), T1, tot); }
    launch_gemm(s, T1, F(218) /*pe_w*/, F(215) /*pe_b*/, nullptr, T2, (long)B * 3136, 96, 48, false, 0);
    launch_ln(s, T2, F(217) /*pe_ln.w*/, F(216) /*pe_ln.b*/, X0, (long)B * 3136, 96);

    // ---- encoder ----
    const int encH[4]  = {56, 28, 14, 7};
    const int encC[4]  = {96, 192, 384, 768};
    const int encNH[4] = {3, 6, 12, 24};
    const int encB0[4] = {93, 122, 151, 180};
    const int encB1[4] = {106, 135, 164, 193};
    const int dLnB[3]  = {119, 148, 177};
    const int dLnW[3]  = {120, 149, 178};
    const int dW[3]    = {121, 150, 179};
    float* Dbuf[3] = {D0, D1, D2};

    for (int i = 0; i < 4; ++i) {
        const int H = encH[i], W = encH[i], C = encC[i];
        const long rows = (long)B * H * W;
        if (i < 3)
            hipMemcpyAsync(Dbuf[i], X0, (size_t)rows * C * sizeof(float),
                           hipMemcpyDeviceToDevice, s);
        swin_block(s, d_in, encB0[i], X0, B, H, W, C, encNH[i], 0, T1, T2, T3);
        swin_block(s, d_in, encB1[i], X0, B, H, W, C, encNH[i], (H > 7) ? 3 : 0, T1, T2, T3);
        if (i < 3) {
            const long tot = rows * C;   // == rows/4 * 4C
            merge_gather_k<<<GRID1D(tot)>>>(X0, T1, B, H, W, C, tot);
            launch_ln(s, T1, F(dLnW[i]), F(dLnB[i]), T1, rows / 4, 4 * C);
            launch_gemm(s, T1, F(dW[i]), nullptr, nullptr, X0, rows / 4, 2 * C, 4 * C, false, 0);
        }
    }

    // ---- bottleneck norm + mem0 + dec0 expand ----
    launch_ln(s, X0, F(211) /*norm.w*/, F(210) /*norm.b*/, X0, (long)B * 49, 768);
    mem_module(s, X0, F(206) /*mem[0]*/, XS, B, 7, 7, 768, 1, T1, T2, T3);
    launch_gemm(s, XS, F(8) /*dec0.w*/, nullptr, nullptr, T1, (long)B * 49, 1536, 768, false, 0);
    { const long tot = (long)B * 196 * 384;
      expand_rearrange_k<<<GRID1D(tot)>>>(T1, X0, B, 7, 7, 768, tot); }
    launch_ln(s, X0, F(7) /*dec0.ln.w*/, F(6) /*dec0.ln.b*/, X0, (long)B * 196, 384);

    // ---- decoder ----
    const int decH[3]   = {14, 28, 56};
    const int decDim[3] = {384, 192, 96};
    const int decNH[3]  = {12, 6, 3};
    const int decB0[3]  = {9, 38, 67};
    const int decB1[3]  = {22, 51, 80};
    const int upLnB[2]  = {35, 64};
    const int upLnW[2]  = {36, 65};
    const int upW[2]    = {37, 66};
    const int cbB[3]    = {0, 2, 4};
    const int cbW[3]    = {1, 3, 5};
    const int memIdx[3] = {207, 208, 209};
    const int wsz[3]    = {2, 4, 8};
    float* Dskip[3] = {D2, D1, D0};

    for (int k = 0; k < 3; ++k) {   // inx = k+1
        const int H = decH[k], W = decH[k], dim = decDim[k];
        const long rows = (long)B * H * W;
        mem_module(s, X0, F(memIdx[k]), XS, B, H, W, dim, wsz[k], T1, T2, T3);
        { const long tot = rows * 2 * dim;
          concat2_k<<<GRID1D(tot)>>>(XS, Dskip[k], T1, dim, tot); }
        launch_gemm(s, T1, F(cbW[k]), F(cbB[k]), nullptr, X0, rows, dim, 2 * dim, false, 0);
        swin_block(s, d_in, decB0[k], X0, B, H, W, dim, decNH[k], 0, T1, T2, T3);
        swin_block(s, d_in, decB1[k], X0, B, H, W, dim, decNH[k], 3, T1, T2, T3);
        if (k < 2) {
            launch_gemm(s, X0, F(upW[k]), nullptr, nullptr, T1, rows, 2 * dim, dim, false, 0);
            const long tot = rows * 2 * dim;   // == 4*rows * dim/2
            expand_rearrange_k<<<GRID1D(tot)>>>(T1, X0, B, H, W, dim, tot);
            launch_ln(s, X0, F(upLnW[k]), F(upLnB[k]), X0, rows * 4, dim / 2);
        }
    }

    // ---- final 4x expand + 1x1 conv ----
    launch_ln(s, X0, F(213) /*norm_up.w*/, F(212) /*norm_up.b*/, X0, (long)B * 3136, 96);
    launch_gemm(s, X0, F(221) /*up_w*/, nullptr, nullptr, BIG1, (long)B * 3136, 1536, 96, false, 0);
    { const long tot = (long)B * 224 * 224 * 96; up_rearrange_k<<<GRID1D(tot)>>>(BIG1, BIG2, tot); }
    launch_ln(s, BIG2, F(220) /*up_ln.w*/, F(219) /*up_ln.b*/, BIG2, (long)B * 224 * 224, 96);
    launch_gemm(s, BIG2, F(214) /*out_w*/, nullptr, nullptr, XS, (long)B * 224 * 224, 3, 96, false, 0);
    { const long tot = (long)B * 3 * 224 * 224;
      out_transpose_k<<<GRID1D(tot)>>>(XS, (float*)d_out, tot); }
}
